// DDSPCore_26817775796342
// MI455X (gfx1250) — matrix operand order repaired and checked by reading, not compiled, not run on hardware
//
#include <hip/hip_runtime.h>
#include <math.h>

// ---- problem constants (match reference) ----
#define SR_F      44100.0f
#define TWO_PI_F  6.28318530717958647692f
constexpr int B_    = 32;
constexpr int T_    = 64000;
constexpr int SEGLEN = 256;              // threads per block, samples per segment
constexpr int NSEG   = T_ / SEGLEN;      // 250, exact

typedef __attribute__((ext_vector_type(16))) _Float16 v16h;
typedef __attribute__((ext_vector_type(8)))  float    v8f;

// -------------------------------------------------------------------------
// Kernel 1: per-segment sums of the phase increment (for the segmented scan)
// -------------------------------------------------------------------------
__global__ void __launch_bounds__(SEGLEN)
seg_sum_kernel(const float* __restrict__ f0, float* __restrict__ segsum) {
    int blk = blockIdx.x;
    int b   = blk / NSEG;
    int seg = blk % NSEG;
    int t   = seg * SEGLEN + threadIdx.x;

    // phase_inc = 2*pi * 20*exp(f0) / SR
    float inc = (TWO_PI_F * 20.0f / SR_F) * __expf(f0[b * T_ + t]);

    __shared__ float red[SEGLEN];
    red[threadIdx.x] = inc;
    __syncthreads();
    for (int off = SEGLEN / 2; off > 0; off >>= 1) {
        if (threadIdx.x < off) red[threadIdx.x] += red[threadIdx.x + off];
        __syncthreads();
    }
    if (threadIdx.x == 0) segsum[b * NSEG + seg] = red[0];
}

// -------------------------------------------------------------------------
// Kernel 2: serial exclusive scan of segment sums (32 rows, 250 segs each)
// -------------------------------------------------------------------------
__global__ void seg_scan_kernel(const float* __restrict__ segsum,
                                float* __restrict__ bases) {
    int b = threadIdx.x;
    if (b < B_) {
        float acc = 0.0f;
        for (int s = 0; s < NSEG; ++s) {
            bases[b * NSEG + s] = acc;
            acc += segsum[b * NSEG + s];
        }
    }
}

// -------------------------------------------------------------------------
// helpers for the WMMA harmonic bank
// -------------------------------------------------------------------------
// Fill 8 consecutive f16 elements of an A-fragment with sin(k*ph) for
// k = k0..k0+7 using one sincos + angle-add + Chebyshev recurrence.
__device__ __forceinline__ void fill_run(v16h& a, int e0, int k0, float ph,
                                         float s1, float c1, float c2) {
    float sk, ck;
    __sincosf((float)k0 * ph, &sk, &ck);
    float sp = sk;                    // sin(k0*ph)
    float sc = sk * c1 + ck * s1;     // sin((k0+1)*ph) via angle addition
    a[e0 + 0] = (_Float16)sp;
    a[e0 + 1] = (_Float16)sc;
#pragma unroll
    for (int j = 2; j < 8; ++j) {     // sin((k+1)ph) = 2cos(ph)sin(k ph) - sin((k-1)ph)
        float sn = c2 * sc - sp;
        a[e0 + j] = (_Float16)sn;
        sp = sc; sc = sn;
    }
}

__device__ __forceinline__ float sel8(const v8f& c, int i) {
    float v = c[0];
    if (i == 1) v = c[1];
    if (i == 2) v = c[2];
    if (i == 3) v = c[3];
    if (i == 4) v = c[4];
    if (i == 5) v = c[5];
    if (i == 6) v = c[6];
    if (i == 7) v = c[7];
    return v;
}

// -------------------------------------------------------------------------
// Kernel 3: main — in-block phase scan, WMMA harmonic bank, mix, peak atomics
// -------------------------------------------------------------------------
__global__ void __launch_bounds__(SEGLEN)
ddsp_main_kernel(const float* __restrict__ f0, const float* __restrict__ loud,
                 const float* __restrict__ mix, const float* __restrict__ noise,
                 const float* __restrict__ bases, float* __restrict__ out,
                 unsigned* __restrict__ peaks) {
    int blk  = blockIdx.x;
    int b    = blk / NSEG;
    int seg  = blk % NSEG;
    int tid  = threadIdx.x;
    int lane = tid & 31;
    int wave = tid >> 5;
    int t    = seg * SEGLEN + tid;
    int gidx = b * T_ + t;

    float inc = (TWO_PI_F * 20.0f / SR_F) * __expf(f0[gidx]);

    // ---- block-wide inclusive scan (Hillis-Steele, double buffered) ----
    __shared__ float sa[SEGLEN], sb[SEGLEN];
    __shared__ float res[SEGLEN];
    __shared__ unsigned bmax;
    if (tid == 0) bmax = 0u;
    sa[tid] = inc;
    __syncthreads();
    float* src = sa;
    float* dst = sb;
    for (int off = 1; off < SEGLEN; off <<= 1) {
        float v = src[tid];
        if (tid >= off) v += src[tid - off];
        dst[tid] = v;
        __syncthreads();
        float* tmp = src; src = dst; dst = tmp;
    }
    float phase = bases[b * NSEG + seg] + src[tid];
    // range-reduce once; harmonics scale the reduced angle
    float phir = phase - TWO_PI_F * floorf(phase * (1.0f / TWO_PI_F));

    // ---- per-lane constant weight fragments (B matrix: w in all 16 cols) ----
    // B layout (16-bit, 32x16, wave32): lanes 0-15 hold K=0..7,16..23, lanes 16-31 K=8..15,24..31
    int hi = (lane >= 16) ? 1 : 0;
    v16h w1, w2;
#pragma unroll
    for (int e = 0; e < 16; ++e) {
        int K  = (e < 8 ? e : e + 8) + (hi ? 8 : 0);
        int h1 = K + 1;                         // harmonics 1..32
        int h2 = K + 33;                        // harmonics 33..64 (61..64 padded 0)
        w1[e] = (_Float16)(1.0f / (float)h1);
        w2[e] = (_Float16)((h2 <= 60) ? (1.0f / (float)h2) : 0.0f);
    }

    // ---- two 16-sample tiles per wave; each tile = 2 chained WMMAs (K=64) ----
    int kh = hi ? 8 : 0;   // A layout: lanes>=16 own K rows shifted by +8 per half
#pragma unroll
    for (int tile = 0; tile < 2; ++tile) {
        int tb = tile * 16;
        // this lane's assigned matrix row M = lane&15 -> wave-local sample tb+M
        float pm = __shfl(phir, tb + (lane & 15), 32);
        float s1, c1;
        __sincosf(pm, &s1, &c1);
        float c2 = 2.0f * c1;

        // A layout (16-bit, 16x32): lane<16 elems 0..7 -> K=0..7, 8..15 -> K=16..23
        //                           lane>=16 elems 0..7 -> K=8..15, 8..15 -> K=24..31
        v16h a1, a2;
        fill_run(a1, 0, 1 + kh, pm, s1, c1, c2);    // harmonics  1..8 (+kh)
        fill_run(a1, 8, 17 + kh, pm, s1, c1, c2);   // harmonics 17..24 (+kh)
        fill_run(a2, 0, 33 + kh, pm, s1, c1, c2);   // harmonics 33..40 (+kh)
        fill_run(a2, 8, 49 + kh, pm, s1, c1, c2);   // harmonics 49..56 (+kh)

        v8f c = {};
        c = __builtin_amdgcn_wmma_f32_16x16x32_f16(false, a1, false, w1,
                                                   (short)0, c, false, false);
        c = __builtin_amdgcn_wmma_f32_16x16x32_f16(false, a2, false, w2,
                                                   (short)0, c, false, false);

        // D layout: VGPR r holds row M=r (lanes 0-15) / M=r+8 (lanes 16-31);
        // all 16 columns are identical, so lanes 0-7 emit rows 0-7 and
        // lanes 16-23 emit rows 8-15.
        if ((lane & 15) < 8) {
            int row = (lane < 16) ? lane : (lane - 8);
            res[wave * 32 + tb + row] = sel8(c, lane & 7);
        }
    }
    __syncthreads();

    // ---- epilogue: mix, store unnormalized audio, track row peak ----
    float hw = res[tid];
    float l  = loud[gidx];
    float m  = mix[gidx];
    float n  = noise[gidx];
    float audio = m * (hw * l) + (1.0f - m) * (n * l);
    out[gidx] = audio;

    atomicMax(&bmax, __float_as_uint(fabsf(audio)));
    __syncthreads();
    if (tid == 0) atomicMax(&peaks[b], bmax);
}

// -------------------------------------------------------------------------
// Kernel 4: normalize by per-row peak
// -------------------------------------------------------------------------
__global__ void __launch_bounds__(256)
normalize_kernel(float* __restrict__ out, const unsigned* __restrict__ peaks) {
    int i = blockIdx.x * blockDim.x + threadIdx.x;
    int b = i / T_;
    float peak = __uint_as_float(peaks[b]);
    out[i] = out[i] / (peak + 1e-6f);
}

// -------------------------------------------------------------------------
extern "C" void kernel_launch(void* const* d_in, const int* in_sizes, int n_in,
                              void* d_out, int out_size, void* d_ws, size_t ws_size,
                              hipStream_t stream) {
    const float* f0    = (const float*)d_in[0];
    const float* loud  = (const float*)d_in[1];
    const float* mix   = (const float*)d_in[2];
    const float* noise = (const float*)d_in[3];
    float* out = (float*)d_out;

    char* ws = (char*)d_ws;
    float*    segsum = (float*)ws;                                      // B*NSEG floats
    float*    bases  = (float*)(ws + (size_t)B_ * NSEG * sizeof(float));// B*NSEG floats
    unsigned* peaks  = (unsigned*)(ws + (size_t)2 * B_ * NSEG * sizeof(float)); // B uints

    hipMemsetAsync(peaks, 0, B_ * sizeof(unsigned), stream);

    seg_sum_kernel<<<B_ * NSEG, SEGLEN, 0, stream>>>(f0, segsum);
    seg_scan_kernel<<<1, 32, 0, stream>>>(segsum, bases);
    ddsp_main_kernel<<<B_ * NSEG, SEGLEN, 0, stream>>>(f0, loud, mix, noise,
                                                       bases, out, peaks);
    normalize_kernel<<<(B_ * T_) / 256, 256, 0, stream>>>(out, peaks);
}
